// TauPolicyNetwork_28492813042260
// MI455X (gfx1250) — matrix operand order repaired and checked by reading, not compiled, not run on hardware
//
#include <hip/hip_runtime.h>
#include <hip/hip_bf16.h>
#include <math.h>

// ---------- types ----------
typedef __bf16 bf16;
typedef __attribute__((ext_vector_type(16))) __bf16 v16bf;
typedef __attribute__((ext_vector_type(8)))  __bf16 v8bf;
typedef __attribute__((ext_vector_type(4)))  __bf16 v4bf;
typedef __attribute__((ext_vector_type(8)))  float  v8f;
typedef __attribute__((ext_vector_type(4)))  unsigned u32x4;
typedef __attribute__((ext_vector_type(8)))  int      i32x8;
typedef __attribute__((ext_vector_type(4)))  int      i32x4;

union V16 { v16bf v; v8bf h[2]; };
union BV8 { v8bf v; bf16 b[8]; unsigned short s[8]; };

static __device__ __forceinline__ bf16 f2bf(float f) { return (bf16)f; }   // native v_cvt (RNE)
static __device__ __forceinline__ float bf2f(bf16 b) { return (float)b; }
static __device__ __forceinline__ v8f vzero8() {
  v8f z = {0.f,0.f,0.f,0.f,0.f,0.f,0.f,0.f};
  return z;
}

#define WMMA_BF16(A,B,C) __builtin_amdgcn_wmma_f32_16x16x32_bf16(false,(A),false,(B),(short)0,(C),false,false)

#define HAS_TDM __has_builtin(__builtin_amdgcn_tensor_load_to_lds)

// ===================================================================
// Tensor Data Mover: load a tile of bf16 rows (row stride 2048B in
// global) into LDS as [z=chunk][y=row][64B data + 16B pad] (80B rows).
//   x: 8 units of 8B (64B), y: rows (stride 256 units), z: chunks
//   (stride 8 units along the row).  chunks==0 -> 2D tile.
// This toolchain exposes the 6-arg builtin:
//   (u32x4 g0, i32x8 g1, i32x4 g2, i32x4 g3, i32x8 g4, i32 cpol)
// ===================================================================
#if HAS_TDM
static __device__ __forceinline__ void tdm_load_tile(
    const void* gsrc, unsigned lds_off, int rows, int chunks, int tensor_rows)
{
  unsigned long long ga = (unsigned long long)(size_t)gsrc;
  u32x4 g0 = { 1u,                                   // count=1 (valid)
               lds_off,                              // lds_addr
               (unsigned)ga,                         // global_addr lo
               (unsigned)((ga >> 32) & 0x01FFFFFFu) | 0x80000000u }; // hi | type=2
  i32x8 g1;
  g1[0] = (3 << 16) | (1 << 20) | (3 << 22) | (3 << 25); // 8B units, pad_en, every 16dw pad 4dw
  g1[1] = (int)(256u << 16);                // tensor_dim0 = 256 units (2048B row)
  g1[2] = (int)((unsigned)tensor_rows << 16); // tensor_dim1
  g1[3] = (int)(8u << 16);                  // tile_dim0 = 8 units (64B)
  g1[4] = rows | (chunks << 16);            // tile_dim1 | tile_dim2 (0 => 2D)
  g1[5] = 256;                              // tensor_dim0_stride = 256 units
  g1[6] = (8 << 16);                        // tensor_dim1_stride = 8 units (64B)
  g1[7] = 0;
  i32x4 g2 = { 64, 0, 0, 0 };               // tensor_dim2 bound (>= chunks)
  i32x4 g3 = { 0, 0, 0, 0 };
  i32x8 g4 = { 0, 0, 0, 0, 0, 0, 0, 0 };    // zero-filled extra group
  __builtin_amdgcn_tensor_load_to_lds(g0, g1, g2, g3, g4, 0);
}
#endif

// ===================================================================
// Generic fused GEMM: C = epi(A[MxK] @ B[KxN]).  64x64 block tile,
// 128 threads (4 waves), each wave 32x32 via 2x2 WMMA 16x16x32 bf16.
// MODE 0: relu(acc+bias) -> Cf
// MODE 1: acc            -> Cf
// MODE 2: tanh(acc+bias)+Res -> Cf and bf16 Cb
// ===================================================================
template<int MODE>
__global__ __launch_bounds__(128) void gemm64_kernel(
    const float* __restrict__ A, const float* __restrict__ Bm,
    const float* __restrict__ bias, const float* __restrict__ Res,
    float* __restrict__ Cf, bf16* __restrict__ Cb,
    int M, int N, int K)
{
  __shared__ bf16 Al[64][40];   // [row][k], pad 8 -> 80B stride
  __shared__ bf16 Bl[64][40];   // [n][k] (transposed on load)

  const int tid  = threadIdx.x, lane = tid & 31, wid = tid >> 5;
  const int m0 = blockIdx.y * 64, n0 = blockIdx.x * 64;
  const int wm = (wid >> 1) * 32, wn = (wid & 1) * 32;

  v8f acc[2][2];
  #pragma unroll
  for (int i = 0; i < 2; ++i)
    #pragma unroll
    for (int j = 0; j < 2; ++j) acc[i][j] = vzero8();

  const int ar = tid >> 1,  akq = (tid & 1) * 16;   // A: 64 rows x 32 k
  const int bk = tid >> 2,  bnq = (tid & 3) * 16;   // B: 32 k x 64 n
  const int half = (lane & 16) ? 8 : 0;             // A element offset
  const int boff = (lane & 16) ? 8 : 0;             // B element offset
  const int rr   = lane & 15;

  for (int k0 = 0; k0 < K; k0 += 32) {
    if (k0 + 32 < K) {   // prefetch next tiles -> global_prefetch_b8
      __builtin_prefetch(A  + (size_t)(m0 + ar) * K + k0 + 32 + akq, 0, 1);
      __builtin_prefetch(Bm + (size_t)(k0 + 32 + bk) * N + n0 + bnq, 0, 1);
    }
    // stage A (fp32 -> bf16, native cvt)
    const float4* ap = (const float4*)(A + (size_t)(m0 + ar) * K + k0 + akq);
    #pragma unroll
    for (int i = 0; i < 4; ++i) {
      float4 f = ap[i];
      v4bf q = { f2bf(f.x), f2bf(f.y), f2bf(f.z), f2bf(f.w) };
      *(v4bf*)&Al[ar][akq + 4 * i] = q;
    }
    // stage B transposed (fp32 -> bf16)
    const float4* bp = (const float4*)(Bm + (size_t)(k0 + bk) * N + n0 + bnq);
    #pragma unroll
    for (int i = 0; i < 4; ++i) {
      float4 f = bp[i];
      Bl[bnq + 4 * i + 0][bk] = f2bf(f.x);
      Bl[bnq + 4 * i + 1][bk] = f2bf(f.y);
      Bl[bnq + 4 * i + 2][bk] = f2bf(f.z);
      Bl[bnq + 4 * i + 3][bk] = f2bf(f.w);
    }
    __syncthreads();

    V16 a[2], b[2];
    #pragma unroll
    for (int i = 0; i < 2; ++i) {
      a[i].h[0] = *(const v8bf*)&Al[wm + i * 16 + rr][half];
      a[i].h[1] = *(const v8bf*)&Al[wm + i * 16 + rr][16 + half];
      b[i].h[0] = *(const v8bf*)&Bl[wn + i * 16 + rr][boff];
      b[i].h[1] = *(const v8bf*)&Bl[wn + i * 16 + rr][16 + boff];
    }
    #pragma unroll
    for (int i = 0; i < 2; ++i)
      #pragma unroll
      for (int j = 0; j < 2; ++j)
        acc[i][j] = WMMA_BF16(a[i].v, b[j].v, acc[i][j]);
    __syncthreads();
  }

  // epilogue: D layout N = lane&15, M = v + (lane>=16 ? 8 : 0)
  #pragma unroll
  for (int i = 0; i < 2; ++i) {
    #pragma unroll
    for (int j = 0; j < 2; ++j) {
      #pragma unroll
      for (int v = 0; v < 8; ++v) {
        int gm = m0 + wm + i * 16 + v + ((lane & 16) ? 8 : 0);
        int gn = n0 + wn + j * 16 + rr;
        size_t idx = (size_t)gm * N + gn;
        float val = acc[i][j][v];
        if (MODE == 0) {
          val = fmaxf(val + bias[gn], 0.0f);
          Cf[idx] = val;
        } else if (MODE == 1) {
          Cf[idx] = val;
        } else {
          val = tanhf(val + bias[gn]) + Res[idx];
          Cf[idx] = val;
          Cb[idx] = f2bf(val);
        }
      }
    }
  }
}

// ===================================================================
// Row squared-norms of corr (fp32): one block per row
// ===================================================================
__global__ __launch_bounds__(128) void norms_kernel(
    const float* __restrict__ corr, float* __restrict__ norms)
{
  __shared__ float red[128];
  const int row = blockIdx.x;
  const float* p = corr + (size_t)row * 1024;
  float s = 0.f;
  for (int k = threadIdx.x; k < 1024; k += 128) { float v = p[k]; s += v * v; }
  red[threadIdx.x] = s;
  __syncthreads();
  for (int off = 64; off > 0; off >>= 1) {
    if (threadIdx.x < off) red[threadIdx.x] += red[threadIdx.x + off];
    __syncthreads();
  }
  if (threadIdx.x == 0) norms[row] = red[0];
}

// ===================================================================
// Hyperbolic self-attention, flash-style.  One block (256 thr, 8 waves)
// per 32 query rows.  X tiles live in K-chunked LDS layout
// [kc][row][32 bf16 + 8 pad] (80B rows), fed by the Tensor Data Mover;
// next j-tile DMA overlaps softmax+AV.  Online softmax over -dist.
// ===================================================================
#define RS 40            // padded row stride within a chunk (bf16 elems)
#define CS (32 * RS)     // chunk stride = 32 rows
#define TS 40            // padded row stride for XjT / W

// fallback loader producing the same chunked layout (rows = 32)
static __device__ __forceinline__ void load_chunked32(
    const bf16* __restrict__ g, bf16* dst, int tid)
{
  const int j = tid & 31, c0 = (tid >> 5) * 4;
  #pragma unroll
  for (int ci = 0; ci < 4; ++ci) {
    const int kc = c0 + ci;
    const v8bf* src = (const v8bf*)(g + (size_t)j * 1024 + kc * 32);
    bf16* d = dst + kc * CS + j * RS;
    #pragma unroll
    for (int s = 0; s < 4; ++s) *(v8bf*)(d + 8 * s) = src[s];
  }
}

__global__ __launch_bounds__(256) void attn_kernel(
    const bf16* __restrict__ Xg, const float* __restrict__ norms,
    const float* __restrict__ cptr, bf16* __restrict__ attn)
{
  extern __shared__ char smem[];
  bf16*  Xm     = (bf16*)smem;                 // 32 chunks x [32][RS]
  bf16*  Xj     = Xm + 32 * CS;                // 32 chunks x [32][RS]
  bf16*  XjT    = Xj + 32 * CS;                // [1024][TS]  (K-major)
  float* S      = (float*)(XjT + 1024 * TS);   // [32][33] Gram / s values
  bf16*  W      = (bf16*)(S + 32 * 33);        // [32][TS] softmax weights
  float* scaleL = (float*)(W + 32 * TS);       // [32]
  float* rowMax = scaleL + 32;                 // [32]
  float* rowL   = rowMax + 32;                 // [32]
  float* nmL    = rowL + 32;                   // [32]
  float* njL    = nmL + 32;                    // [32]

  const int tid = threadIdx.x, lane = tid & 31, wid = tid >> 5;
  const int m0 = blockIdx.x * 32;
  const float cc = fmaxf(cptr[0], 1e-6f);
  const float inv_sqrt_cc = 1.0f / sqrtf(cc);

#if HAS_TDM
  const unsigned xm_off = (unsigned)(size_t)(void*)Xm;
  const unsigned xj_off = (unsigned)(size_t)(void*)Xj;
  if (wid == 0) {
    tdm_load_tile(Xg + (size_t)m0 * 1024, xm_off, 32, 32, 8192); // resident queries
    tdm_load_tile(Xg,                     xj_off, 32, 32, 8192); // first key tile
  }
#else
  load_chunked32(Xg + (size_t)m0 * 1024, Xm, tid);
#endif
  if (tid < 32) {
    nmL[tid]    = norms[m0 + tid];
    rowMax[tid] = -3.0e38f;
    rowL[tid]   = 0.0f;
  }

  v8f acc[2][8];
  #pragma unroll
  for (int i = 0; i < 2; ++i)
    #pragma unroll
    for (int d = 0; d < 8; ++d) acc[i][d] = vzero8();

  const int half = (lane & 16) ? 8 : 0;
  const int boff = (lane & 16) ? 8 : 0;
  const int rr   = lane & 15;

  for (int jb = 0; jb < 8192 / 32; ++jb) {
    const int j0 = jb * 32;
#if HAS_TDM
    if (wid == 0) __builtin_amdgcn_s_wait_tensorcnt((short)0);
    __syncthreads();      // Xj(jb) visible; previous AV done (XjT/W free)
#else
    __syncthreads();      // previous AV / Gram done
    load_chunked32(Xg + (size_t)j0 * 1024, Xj, tid);
    __syncthreads();
#endif

    // build K-major copy XjT[d][j] from chunked Xj (LDS -> LDS repack)
    {
      const int j  = tid & 31;
      const int c0 = (tid >> 5) * 4;
      #pragma unroll
      for (int ci = 0; ci < 4; ++ci) {
        const int kc = c0 + ci;
        const bf16* src = Xj + kc * CS + j * RS;
        BV8 r[4];
        #pragma unroll
        for (int s = 0; s < 4; ++s) r[s].v = *(const v8bf*)(src + 8 * s);
        #pragma unroll
        for (int s = 0; s < 4; ++s)
          #pragma unroll
          for (int e = 0; e < 8; ++e)
            XjT[(size_t)(kc * 32 + 8 * s + e) * TS + j] = r[s].b[e];
      }
    }
    if (tid < 32) njL[tid] = norms[j0 + tid];

    // Gram: waves 0..3, each one 16x16 tile of S^T (A = Xj rows, B = Xm rows)
    if (wid < 4) {
      const int jt = (wid & 1) * 16, mt = (wid >> 1) * 16;
      v8f g = vzero8();
      #pragma unroll 4
      for (int kc = 0; kc < 32; ++kc) {
        const bf16* xa = Xj + kc * CS + (jt + rr) * RS;
        const bf16* xb = Xm + kc * CS + (mt + rr) * RS;
        V16 a, b;
        a.h[0] = *(const v8bf*)(xa + half);
        a.h[1] = *(const v8bf*)(xa + 16 + half);
        b.h[0] = *(const v8bf*)(xb + boff);
        b.h[1] = *(const v8bf*)(xb + 16 + boff);
        g = WMMA_BF16(a.v, b.v, g);
      }
      #pragma unroll
      for (int v = 0; v < 8; ++v) {
        int jr = jt + v + ((lane & 16) ? 8 : 0);
        S[(mt + rr) * 33 + jr] = g[v];   // store transposed -> S[m][j]
      }
    }
    __syncthreads();      // Gram + repack done; Xj buffer free

#if HAS_TDM
    if (wid == 0 && jb + 1 < 8192 / 32)   // async DMA of next tile overlaps softmax+AV
      tdm_load_tile(Xg + (size_t)(j0 + 32) * 1024, xj_off, 32, 32, 8192);
#endif

    // online softmax over -dist (one thread per query row)
    if (tid < 32) {
      const int m = tid;
      const float nm = nmL[m];
      const float tm = fmaxf(1.0f - cc * nm, 1e-6f);
      float tmax = -3.0e38f;
      for (int j = 0; j < 32; ++j) {
        float g  = S[m * 33 + j];
        float nj = njL[j];
        float d2 = fmaxf(nm + nj - 2.0f * g, 0.0f);
        float tj = fmaxf(1.0f - cc * nj, 1e-6f);
        float arg = fmaxf(1.0f + (2.0f * cc) * d2 / (tm * tj), 1.0f + 1e-6f) + 1e-6f;
        float dist = logf(arg + sqrtf(arg * arg - 1.0f)) * inv_sqrt_cc;  // arccosh
        float sv = -dist;
        S[m * 33 + j] = sv;               // reuse S as s-value scratch
        tmax = fmaxf(tmax, sv);
      }
      float om   = rowMax[m];
      float nmax = fmaxf(om, tmax);
      float sc   = expf(om - nmax);
      float l    = rowL[m] * sc;
      for (int j = 0; j < 32; ++j) {
        float w = expf(S[m * 33 + j] - nmax);
        l += w;
        W[m * TS + j] = f2bf(w);
      }
      rowMax[m] = nmax;
      rowL[m]   = l;
      scaleL[m] = sc;
    }
    __syncthreads();

    // AV: acc = acc*scale + W @ Xj   (each wave owns 128 feature columns)
    #pragma unroll
    for (int i = 0; i < 2; ++i)
      #pragma unroll
      for (int v = 0; v < 8; ++v) {
        float sc = scaleL[i * 16 + v + ((lane & 16) ? 8 : 0)];
        #pragma unroll
        for (int dt = 0; dt < 8; ++dt) acc[i][dt][v] *= sc;
      }
    #pragma unroll
    for (int i = 0; i < 2; ++i) {
      V16 a;
      a.h[0] = *(const v8bf*)&W[(i * 16 + rr) * TS + half];
      a.h[1] = *(const v8bf*)&W[(i * 16 + rr) * TS + 16 + half];
      #pragma unroll
      for (int dt = 0; dt < 8; ++dt) {
        V16 b;
        const int d = wid * 128 + dt * 16 + rr;
        b.h[0] = *(const v8bf*)&XjT[(size_t)d * TS + boff];
        b.h[1] = *(const v8bf*)&XjT[(size_t)d * TS + 16 + boff];
        acc[i][dt] = WMMA_BF16(a.v, b.v, acc[i][dt]);
      }
    }
  }

  __syncthreads();
  // finalize: divide by softmax denominator, emit bf16 attn
  #pragma unroll
  for (int i = 0; i < 2; ++i) {
    float rl[8];
    #pragma unroll
    for (int v = 0; v < 8; ++v)
      rl[v] = 1.0f / rowL[i * 16 + v + ((lane & 16) ? 8 : 0)];
    #pragma unroll
    for (int dt = 0; dt < 8; ++dt) {
      const int d = wid * 128 + dt * 16 + rr;
      #pragma unroll
      for (int v = 0; v < 8; ++v) {
        int gm = m0 + i * 16 + v + ((lane & 16) ? 8 : 0);
        attn[(size_t)gm * 1024 + d] = f2bf(acc[i][dt][v] * rl[v]);
      }
    }
  }
}

// ===================================================================
// Actor head: logits = attn @ W_actor + b_actor, then row softmax.
// 128 threads (4 waves), 64 rows per block; A tile arrives via TDM.
// ===================================================================
__global__ __launch_bounds__(128) void heads_kernel(
    const bf16* __restrict__ attn, const float* __restrict__ Wa,
    const float* __restrict__ ba, float* __restrict__ logits,
    float* __restrict__ probs)
{
  __shared__ bf16  Al[64][40];
  __shared__ bf16  Bl[64][40];
  __shared__ float Ls[64][65];
  const int tid = threadIdx.x, lane = tid & 31, wid = tid >> 5;
  const int m0 = blockIdx.x * 64;
  v8f acc[4];
  #pragma unroll
  for (int j = 0; j < 4; ++j) acc[j] = vzero8();

  const int bk = tid >> 2, bnq = (tid & 3) * 16;
  const int half = (lane & 16) ? 8 : 0;
  const int boff = (lane & 16) ? 8 : 0;
  const int rr   = lane & 15;

#if HAS_TDM
  const unsigned al_off = (unsigned)(size_t)(void*)&Al[0][0];
  if (wid == 0) tdm_load_tile(attn + (size_t)m0 * 1024, al_off, 64, 0, 8192);
#else
  const int ar = tid >> 1, akq = (tid & 1) * 16;
#endif

  for (int k0 = 0; k0 < 1024; k0 += 32) {
#if !HAS_TDM
    *(v8bf*)&Al[ar][akq]     = *(const v8bf*)(attn + (size_t)(m0 + ar) * 1024 + k0 + akq);
    *(v8bf*)&Al[ar][akq + 8] = *(const v8bf*)(attn + (size_t)(m0 + ar) * 1024 + k0 + akq + 8);
#endif
    const float4* bp = (const float4*)(Wa + (size_t)(k0 + bk) * 64 + bnq);
    #pragma unroll
    for (int i = 0; i < 4; ++i) {
      float4 f = bp[i];
      Bl[bnq + 4 * i + 0][bk] = f2bf(f.x);
      Bl[bnq + 4 * i + 1][bk] = f2bf(f.y);
      Bl[bnq + 4 * i + 2][bk] = f2bf(f.z);
      Bl[bnq + 4 * i + 3][bk] = f2bf(f.w);
    }
#if HAS_TDM
    if (wid == 0) __builtin_amdgcn_s_wait_tensorcnt((short)0);
#endif
    __syncthreads();
    V16 a;
    a.h[0] = *(const v8bf*)&Al[wid * 16 + rr][half];
    a.h[1] = *(const v8bf*)&Al[wid * 16 + rr][16 + half];
    #pragma unroll
    for (int j = 0; j < 4; ++j) {
      V16 b;
      b.h[0] = *(const v8bf*)&Bl[j * 16 + rr][boff];
      b.h[1] = *(const v8bf*)&Bl[j * 16 + rr][16 + boff];
      acc[j] = WMMA_BF16(a.v, b.v, acc[j]);
    }
    __syncthreads();
#if HAS_TDM
    if (wid == 0 && k0 + 32 < 1024)   // next A tile overlaps B staging of next step
      tdm_load_tile(attn + (size_t)m0 * 1024 + k0 + 32, al_off, 64, 0, 8192);
#endif
  }

  #pragma unroll
  for (int j = 0; j < 4; ++j)
    #pragma unroll
    for (int v = 0; v < 8; ++v) {
      int r = wid * 16 + v + ((lane & 16) ? 8 : 0);
      int n = j * 16 + rr;
      Ls[r][n] = acc[j][v] + ba[n];
    }
  __syncthreads();

  if (tid < 64) {
    const int r = tid;
    float mx = -3.0e38f;
    for (int n = 0; n < 64; ++n) mx = fmaxf(mx, Ls[r][n]);
    float sum = 0.f;
    for (int n = 0; n < 64; ++n) sum += expf(Ls[r][n] - mx);
    float inv = 1.0f / sum;
    for (int n = 0; n < 64; ++n) {
      float lv = Ls[r][n];
      size_t o = (size_t)(m0 + r) * 64 + n;
      logits[o] = lv;
      probs[o]  = expf(lv - mx) * inv;
    }
  }
}

// ===================================================================
// Risk head: sigmoid(attn @ W_risk + b_risk).  One wave per row.
// ===================================================================
__global__ __launch_bounds__(256) void risk_kernel(
    const bf16* __restrict__ attn, const float* __restrict__ Wr,
    const float* __restrict__ br, float* __restrict__ risk)
{
  const int lane = threadIdx.x & 31, wid = threadIdx.x >> 5;
  const int row = blockIdx.x * 8 + wid;
  float s = 0.f;
  for (int k = lane; k < 1024; k += 32)
    s += bf2f(attn[(size_t)row * 1024 + k]) * Wr[k];
  #pragma unroll
  for (int off = 16; off > 0; off >>= 1) s += __shfl_xor(s, off);
  if (lane == 0) risk[row] = 1.0f / (1.0f + expf(-(s + br[0])));
}

// ===================================================================
// launcher
// ===================================================================
extern "C" void kernel_launch(void* const* d_in, const int* in_sizes, int n_in,
                              void* d_out, int out_size, void* d_ws, size_t ws_size,
                              hipStream_t stream)
{
  (void)in_sizes; (void)n_in; (void)out_size; (void)ws_size;
  const float* state   = (const float*)d_in[0];
  const float* W_enc   = (const float*)d_in[1];
  const float* b_enc   = (const float*)d_in[2];
  const float* cov     = (const float*)d_in[3];
  const float* W_phase = (const float*)d_in[4];
  const float* b_phase = (const float*)d_in[5];
  const float* cptr    = (const float*)d_in[6];
  const float* W_actor = (const float*)d_in[7];
  const float* b_actor = (const float*)d_in[8];
  const float* W_risk  = (const float*)d_in[9];
  const float* b_risk  = (const float*)d_in[10];

  const int Bsz = 8192, IN = 512, HID = 1024, ACT = 64;

  char* w = (char*)d_ws;
  float* enc_f  = (float*)w;  w += (size_t)Bsz * HID * 4;
  float* tmp_f  = (float*)w;  w += (size_t)Bsz * HID * 4;
  float* corr_f = (float*)w;  w += (size_t)Bsz * HID * 4;
  bf16*  corr_b = (bf16*)w;   w += (size_t)Bsz * HID * 2;
  bf16*  attn_b = (bf16*)w;   w += (size_t)Bsz * HID * 2;
  float* norms  = (float*)w;  w += (size_t)Bsz * 4;

  float* out_logits = (float*)d_out;
  float* out_probs  = out_logits + (size_t)Bsz * ACT;
  float* out_risk   = out_probs + (size_t)Bsz * ACT;

  gemm64_kernel<0><<<dim3(HID / 64, Bsz / 64), 128, 0, stream>>>(
      state, W_enc, b_enc, nullptr, enc_f, nullptr, Bsz, HID, IN);
  gemm64_kernel<1><<<dim3(HID / 64, Bsz / 64), 128, 0, stream>>>(
      enc_f, cov, nullptr, nullptr, tmp_f, nullptr, Bsz, HID, HID);
  gemm64_kernel<2><<<dim3(HID / 64, Bsz / 64), 128, 0, stream>>>(
      tmp_f, W_phase, b_phase, enc_f, corr_f, corr_b, Bsz, HID, HID);
  norms_kernel<<<Bsz, 128, 0, stream>>>(corr_f, norms);

  // dynamic LDS: Xm + Xj (chunked) + XjT + W (bf16) and S + row state (f32)
  const size_t smem = (size_t)(32 * CS + 32 * CS + 1024 * TS + 32 * TS) * 2
                    + (size_t)(32 * 33 + 5 * 32) * 4;   // = 253,184 B (< 320 KB WGP)
  attn_kernel<<<Bsz / 32, 256, smem, stream>>>(corr_b, norms, cptr, attn_b);

  heads_kernel<<<Bsz / 64, 128, 0, stream>>>(attn_b, W_actor, b_actor, out_logits, out_probs);
  risk_kernel<<<Bsz / 8, 256, 0, stream>>>(attn_b, W_risk, b_risk, out_risk);
}
